// BlackBoxModel_24489903521937
// MI455X (gfx1250) — matrix operand order repaired and checked by reading, not compiled, not run on hardware
//
#include <hip/hip_runtime.h>
#include <math.h>

typedef __attribute__((ext_vector_type(16))) _Float16 v16h;
typedef __attribute__((ext_vector_type(8)))  float    v8f;
typedef _Float16 f16;

namespace {
constexpr int kNY = 8, kNU = 4;
constexpr int kT  = 256;
constexpr int kH  = 128;
constexpr int kDS = 56;   // D_STATE
constexpr int kDI = 60;   // D_IN
constexpr int WAVES = 4;  // waves per block
constexpr int ROWS  = 16; // batch rows per wave (WMMA M)
constexpr int XS = 72;    // x row stride in halves (144B, 16B-aligned, bank-spread)
constexpr int HS = 136;   // h / W2^T row stride in halves (272B, 16B-aligned)
}

// Branch-free tanh on the hardware transcendental path:
// tanh(x) = 1 - 2 / (1 + e^{2x}) = 1 - 2*rcp(1 + exp2(x * 2*log2(e)))
// x->+inf: exp2->inf, rcp->0, result 1.  x->-inf: exp2->0, result -1.  NaN propagates.
__device__ __forceinline__ float fast_tanh(float x) {
  const float e = __builtin_amdgcn_exp2f(x * 2.885390081777926814f);
  return __builtin_fmaf(-2.0f, __builtin_amdgcn_rcpf(1.0f + e), 1.0f);
}

// A-fragment (16x32 f16): lane holds M=lane%16, K = e + 16*(lane/16) (contiguous, matches B). Two 16B LDS reads.
__device__ __forceinline__ v16h ldA(const f16* row, int c, int hi) {
  v16h r;
  const uint4* p0 = (const uint4*)(row + 32 * c + 16 * hi);
  const uint4* p1 = (const uint4*)(row + 32 * c + 16 * hi + 8);
  ((uint4*)&r)[0] = *p0;
  ((uint4*)&r)[1] = *p1;
  return r;
}

// B-fragment (32x16 f16): lane holds N=lane%16, K = e + 16*(lane/16), contiguous in N-major W^T.
__device__ __forceinline__ v16h ldB(const f16* row, int c, int hi) {
  v16h r;
  const uint4* p = (const uint4*)(row + 32 * c + 16 * hi);
  ((uint4*)&r)[0] = p[0];
  ((uint4*)&r)[1] = p[1];
  return r;
}

__device__ __forceinline__ v8f wmma_f16(v16h a, v16h b, v8f c) {
  return __builtin_amdgcn_wmma_f32_16x16x32_f16(false, a, false, b, (short)0, c, false, false);
}

__global__ __launch_bounds__(WAVES * 32, 1)
void narx_rnn_wmma(const float* __restrict__ useq, const float* __restrict__ yz0,
                   const float* __restrict__ W1,   const float* __restrict__ b1,
                   const float* __restrict__ W2,   const float* __restrict__ b2,
                   const float* __restrict__ W3,   const float* __restrict__ b3,
                   float* __restrict__ out)
{
  __shared__ f16 w2T[kH][HS];                // W2^T [n][k], shared by all waves
  __shared__ f16 xbuf[WAVES][ROWS][XS];      // per-wave state x (padded to 64 cols)
  __shared__ f16 hbuf[WAVES][ROWS][HS];      // per-wave hidden activations

  const int tid  = threadIdx.x;
  const int wave = tid >> 5;
  const int lane = tid & 31;
  const int lo   = lane & 15;
  const int hi   = lane >> 4;
  const int b0   = blockIdx.x * (WAVES * ROWS) + wave * ROWS;

  // --- cooperative: W2^T (f32 -> f16) into LDS ---
  for (int idx = tid; idx < kH * kH; idx += blockDim.x) {
    const int k = idx >> 7, n = idx & 127;
    w2T[n][k] = (f16)W2[idx];
  }

  // --- loop-invariant register weight fragments: W1 (2 K-chunks x 8 N-tiles) ---
  v16h w1f[8][2];
#pragma unroll
  for (int j = 0; j < 8; ++j)
#pragma unroll
    for (int c = 0; c < 2; ++c) {
      v16h f;
#pragma unroll
      for (int e = 0; e < 16; ++e) {
        const int k = 32 * c + 16 * hi + e;
        f[e] = (k < kDI) ? (f16)W1[k * kH + (16 * j + lo)] : (f16)0.0f;
      }
      w1f[j][c] = f;
    }
  // W3 (4 K-chunks, N=8 padded to 16 with zeros)
  v16h w3f[4];
#pragma unroll
  for (int c = 0; c < 4; ++c) {
    v16h f;
#pragma unroll
    for (int e = 0; e < 16; ++e) {
      const int k = 32 * c + 16 * hi + e;
      f[e] = (lo < kNY) ? (f16)W3[k * kNY + lo] : (f16)0.0f;
    }
    w3f[c] = f;
  }
  float b1v[8], b2v[8];
#pragma unroll
  for (int j = 0; j < 8; ++j) { b1v[j] = b1[16 * j + lo]; b2v[j] = b2[16 * j + lo]; }
  const float b3v = (lo < kNY) ? b3[lo] : 0.0f;

  f16 (*X)[XS]  = xbuf[wave];
  f16 (*Hm)[HS] = hbuf[wave];

  // --- state init (f16) + ys[:,0] from yz0 (f32, exact) ---
  if (lane < ROWS) {
    const int row = b0 + lane;
#pragma unroll
    for (int k = 0; k < kDS; ++k) X[lane][k] = (f16)yz0[(long)row * kDS + k];
#pragma unroll
    for (int k = kDS; k < 64; ++k) X[lane][k] = (f16)0.0f;
#pragma unroll
    for (int n = 0; n < kNY; ++n)
      out[(long)row * kT * kNY + n] = yz0[(long)row * kDS + n];
  }
  __syncthreads();

#pragma unroll 1
  for (int t = 0; t < kT; ++t) {
    // u(t): append to x[56:60] (also reused for state slot [52:56] after shift)
    float ux = 0.f, uy = 0.f, uz = 0.f, uw = 0.f;
    if (lane < ROWS) {
      const float4 u4 = *(const float4*)(useq + ((long)(b0 + lane) * kT + t) * kNU);
      ux = u4.x; uy = u4.y; uz = u4.z; uw = u4.w;
      X[lane][56] = (f16)ux; X[lane][57] = (f16)uy;
      X[lane][58] = (f16)uz; X[lane][59] = (f16)uw;
    }
    __builtin_amdgcn_wave_barrier();

    // ---- layer 1: h1 = tanh(x @ W1 + b1), x is 16x64 (padded) ----
    const f16* xr = X[lo];
    const v16h a0 = ldA(xr, 0, hi);
    const v16h a1 = ldA(xr, 1, hi);
#pragma unroll
    for (int j = 0; j < 8; ++j) {
      v8f acc;
#pragma unroll
      for (int r = 0; r < 8; ++r) acc[r] = b1v[j];
      acc = wmma_f16(a0, w1f[j][0], acc);
      acc = wmma_f16(a1, w1f[j][1], acc);
#pragma unroll
      for (int r = 0; r < 8; ++r)
        Hm[r + 8 * hi][16 * j + lo] = (f16)fast_tanh(acc[r]);
    }
    __builtin_amdgcn_wave_barrier();

    // ---- layer 2: h2 = tanh(h1 @ W2 + b2) ----
    // All four A-fragments of h1 are register-resident before any h2 store,
    // so each tile's result can be stored immediately (no acc2[] buffering).
    const f16* hr = Hm[lo];
    v16h ha[4];
#pragma unroll
    for (int c = 0; c < 4; ++c) ha[c] = ldA(hr, c, hi);
    __builtin_amdgcn_wave_barrier();
#pragma unroll
    for (int j = 0; j < 8; ++j) {
      v8f acc;
#pragma unroll
      for (int r = 0; r < 8; ++r) acc[r] = b2v[j];
#pragma unroll
      for (int c = 0; c < 4; ++c)
        acc = wmma_f16(ha[c], ldB(&w2T[16 * j + lo][0], c, hi), acc);
#pragma unroll
      for (int r = 0; r < 8; ++r)
        Hm[r + 8 * hi][16 * j + lo] = (f16)fast_tanh(acc[r]);
    }
    __builtin_amdgcn_wave_barrier();

    // ---- layer 3: yplus = h2 @ W3 + b3 (N padded 8->16) ----
    v16h ga[4];
#pragma unroll
    for (int c = 0; c < 4; ++c) ga[c] = ldA(hr, c, hi);
    v8f accY;
#pragma unroll
    for (int r = 0; r < 8; ++r) accY[r] = b3v;
#pragma unroll
    for (int c = 0; c < 4; ++c) accY = wmma_f16(ga[c], w3f[c], accY);

    // ys[:, t+1] = yplus, written in f32 directly from the accumulator
    if ((t + 1) < kT && lo < kNY) {
#pragma unroll
      for (int r = 0; r < 8; ++r)
        out[((long)(b0 + r + 8 * hi) * kT + (t + 1)) * kNY + lo] = accY[r];
    }

    // ---- state shift (wave-synchronous, same-wave LDS ops are in-order) ----
    // new = [yplus(0:8), old[16:40]->(8:32), old[0:8]->(32:40), old[44:56]->(40:52), u->(52:56)]
    f16 tmp[kDS];
    if (lane < ROWS) {
#pragma unroll
      for (int k = 0; k < kDS; ++k) tmp[k] = X[lane][k];
    }
    __builtin_amdgcn_wave_barrier();
    if (lane < ROWS) {
#pragma unroll
      for (int i = 0; i < 24; ++i) X[lane][8 + i]  = tmp[16 + i];
#pragma unroll
      for (int i = 0; i < 8; ++i)  X[lane][32 + i] = tmp[i];
#pragma unroll
      for (int i = 0; i < 12; ++i) X[lane][40 + i] = tmp[44 + i];
      X[lane][52] = (f16)ux; X[lane][53] = (f16)uy;
      X[lane][54] = (f16)uz; X[lane][55] = (f16)uw;
    }
    if (lo < kNY) {
#pragma unroll
      for (int r = 0; r < 8; ++r) X[r + 8 * hi][lo] = (f16)accY[r];
    }
    __builtin_amdgcn_wave_barrier();
  }
}

extern "C" void kernel_launch(void* const* d_in, const int* in_sizes, int n_in,
                              void* d_out, int out_size, void* d_ws, size_t ws_size,
                              hipStream_t stream) {
  (void)n_in; (void)out_size; (void)d_ws; (void)ws_size;
  const float* useq = (const float*)d_in[0];
  const float* yz0  = (const float*)d_in[1];
  const float* W1   = (const float*)d_in[2];
  const float* b1   = (const float*)d_in[3];
  const float* W2   = (const float*)d_in[4];
  const float* b2   = (const float*)d_in[5];
  const float* W3   = (const float*)d_in[6];
  const float* b3   = (const float*)d_in[7];
  float* out = (float*)d_out;

  const int B = in_sizes[1] / kDS;             // yz0 is (B, 56); B = 4096
  const int blocks = B / (WAVES * ROWS);       // 64 rows per block
  narx_rnn_wmma<<<dim3(blocks), dim3(WAVES * 32), 0, stream>>>(
      useq, yz0, W1, b1, W2, b2, W3, b3, out);
}
